// Attention_46325517255302
// MI455X (gfx1250) — matrix operand order repaired and checked by reading, not compiled, not run on hardware
//
#include <hip/hip_runtime.h>
#include <hip/hip_bf16.h>

// ---------------------------------------------------------------------------
// Attention block for MI455X (gfx1250, wave32):
//   WMMA bf16 matmuls + TDM tensor_load_to_lds (GEMM) + async-LDS (flash).
// B=4, N=2048, C=768, H=12, D=64
// ---------------------------------------------------------------------------

#define BB 4
#define NN 2048
#define CC 768
#define HH 12
#define DD 64

typedef __attribute__((ext_vector_type(16))) __bf16          v16bf;
typedef __attribute__((ext_vector_type(8)))  float           v8f;
typedef __attribute__((ext_vector_type(8)))  unsigned int    v8u;
typedef __attribute__((ext_vector_type(4)))  unsigned int    v4u;
typedef __attribute__((ext_vector_type(8)))  int             v8i;
typedef __attribute__((ext_vector_type(4)))  int             v4i;

static __device__ __forceinline__ unsigned short f2bf(float f) {
    unsigned int u = __builtin_bit_cast(unsigned int, f);
    unsigned int r = u + 0x7FFFu + ((u >> 16) & 1u);   // round-to-nearest-even
    return (unsigned short)(r >> 16);
}

static __device__ __forceinline__ v8f vzero() {
    v8f z;
#pragma unroll
    for (int i = 0; i < 8; ++i) z[i] = 0.0f;
    return z;
}

// Async 16-byte copy global -> LDS, tracked by ASYNCcnt.
static __device__ __forceinline__ void async_cp16(unsigned lds_addr, const void* gaddr) {
    asm volatile("global_load_async_to_lds_b128 %0, %1, off"
                 :: "v"(lds_addr), "v"(gaddr) : "memory");
}
#define S_WAIT_ASYNC0 asm volatile("s_wait_asynccnt 0x0" ::: "memory")
#define S_WAIT_ASYNC8 asm volatile("s_wait_asynccnt 0x8" ::: "memory")

// ---------------------------------------------------------------------------
// TDM: DMA a 2D bf16 tile [rows x 32] (row stride = strideElems) from global
// into LDS, with hardware padding of 4 DWORDs after every 16 DWORDs so the
// landed LDS row stride is 40 elements (= GLDA below).
// D# packing per CDNA5 ISA ch.8 (group0: count/lds/global/type; group1:
// data_size, pad, dims, strides). 2D -> groups 2/3 zero.
// ---------------------------------------------------------------------------
static __device__ __forceinline__ void tdm_load_2d(unsigned lds_addr,
                                                   const void* gaddr,
                                                   int rows, int strideElems) {
    unsigned long long ga = (unsigned long long)gaddr;
    v4u g0;
    g0[0] = 1u;                                   // count=1 (valid user D#)
    g0[1] = lds_addr;                             // LDS byte address
    g0[2] = (unsigned)ga;                         // global_addr[31:0]
    g0[3] = (unsigned)(ga >> 32) | (2u << 30);    // global_addr[56:32] | type=2
    v8i g1;
    g1[0] = (int)((1u << 16)        // data_size = 1 -> 2 bytes
                | (1u << 20)        // pad_enable
                | (3u << 22)        // pad_interval: every 16 DWORDs (64 B row)
                | (3u << 25));      // pad_amount: 4 DWORDs (8 elements)
    g1[1] = (int)(32u << 16);                         // tensor_dim0[15:0]=32
    g1[2] = (int)(((unsigned)rows & 0xFFFFu) << 16);  // tensor_dim1[15:0]=rows
    g1[3] = (int)(32u << 16);                         // tile_dim0 = 32
    g1[4] = (int)(rows & 0xFFFF);                     // tile_dim1 = rows
    g1[5] = strideElems;                              // tensor_dim0_stride
    g1[6] = 0;
    g1[7] = 0;
    v4i z4 = {0, 0, 0, 0};
#if __clang_major__ >= 23
    v8i z8 = {0, 0, 0, 0, 0, 0, 0, 0};
    __builtin_amdgcn_tensor_load_to_lds(g0, g1, z4, z4, z8, 0);
#else
    __builtin_amdgcn_tensor_load_to_lds(g0, g1, z4, z4, 0);
#endif
}

// A fragment (16xK tile, K-chunk of 32, bf16) from row-major [row][k].
// ISA layout: lanes 0-15 -> M=lane, K base 0; lanes 16-31 -> M=lane-16,
// K base +8; VGPR v<4: K=2v,2v+1 ; v>=4: K=16+2(v-4),+1.
static __device__ __forceinline__ v16bf load_a(const unsigned short* p, size_t ld,
                                               int kb, int lane) {
    int m = lane & 15, half = lane >> 4;
    const unsigned short* rp = p + (size_t)m * ld + kb + half * 8;
    v8u u;
#pragma unroll
    for (int v = 0; v < 8; ++v) {
        int k = ((v & 4) << 2) + ((v & 3) << 1);
        u[v] = *(const unsigned int*)(rp + k);
    }
    return __builtin_bit_cast(v16bf, u);
}

// B fragment (Kx16) where memory is row-major [n][k] (K contiguous).
// ISA layout: lanes 0-15 -> N=lane, K base 0; lanes 16-31 -> N=lane-16,
// K base +8; VGPR v<4: K=2v,2v+1 ; v>=4: K=16+2(v-4),+1.
static __device__ __forceinline__ v16bf load_bt(const unsigned short* p, size_t ld,
                                                int kb, int lane) {
    int n = lane & 15, half = lane >> 4;
    const unsigned short* rp = p + (size_t)n * ld + kb + half * 8;
    v8u u;
#pragma unroll
    for (int v = 0; v < 8; ++v) u[v] = *(const unsigned int*)(rp + ((v & 4) << 2) + ((v & 3) << 1));
    return __builtin_bit_cast(v16bf, u);
}

static __device__ __forceinline__ v8f wmma_bf16(v16bf a, v16bf b, v8f c) {
    return __builtin_amdgcn_wmma_f32_16x16x32_bf16(false, a, false, b,
                                                   (short)0, c, false, false);
}

// ---------------------------------------------------------------------------
// Kernel 1: fp32 -> bf16 convert
// ---------------------------------------------------------------------------
__global__ void cvt_bf16(const float* __restrict__ src,
                         unsigned short* __restrict__ dst, int n) {
    int i = blockIdx.x * blockDim.x + threadIdx.x;
    if (i < n) dst[i] = f2bf(src[i]);
}

// ---------------------------------------------------------------------------
// Kernel 2/5: C[M,Nc] = A[M,K] * B[Nc,K]^T (+bias), bf16 in, fp32 out.
// Block = 128 threads = 4 waves. Block tile 128x64, wave tile 32x64.
// Double-buffered TDM staging: wave 0 issues two tensor_load_to_lds per
// stage (A 128x32, B 64x32, hardware-padded to 40-element LDS rows) and
// gates with s_wait_tensorcnt; barrier publishes to the other waves.
// ---------------------------------------------------------------------------
#define GLDA 40
#define GB_OFF (128 * GLDA)          // B tile offset (elements) inside a buffer
#define GBUF_ELTS (128 * GLDA + 64 * GLDA)

__global__ __launch_bounds__(128) void gemm_bf16(
        const unsigned short* __restrict__ A,
        const unsigned short* __restrict__ B,
        float* __restrict__ C,
        const float* __restrict__ bias,
        int M, int Nc, int K) {
    __shared__ unsigned short sm[2][GBUF_ELTS];

    int tid  = threadIdx.x;
    int lane = tid & 31;
    int wave = tid >> 5;
    int row0 = blockIdx.y * 128;
    int col0 = blockIdx.x * 64;

    const unsigned short* Ag = A + (size_t)row0 * K;
    const unsigned short* Bg = B + (size_t)col0 * K;

    auto issue = [&](int kb, int buf) {
        if (wave == 0) {
            unsigned base = (unsigned)(size_t)&sm[buf][0];
            tdm_load_2d(base,               Ag + kb, 128, K);
            tdm_load_2d(base + GB_OFF * 2,  Bg + kb, 64,  K);
        }
    };

    v8f acc[8];
#pragma unroll
    for (int t = 0; t < 8; ++t) acc[t] = vzero();

    int nk = K / 32;
    issue(0, 0);
    for (int st = 0; st < nk; ++st) {
        int buf = st & 1;
        if (st + 1 < nk) issue((st + 1) * 32, buf ^ 1);
        if (wave == 0) {
            if (st + 1 < nk) __builtin_amdgcn_s_wait_tensorcnt(2);
            else             __builtin_amdgcn_s_wait_tensorcnt(0);
        }
        __syncthreads();

        const unsigned short* As = &sm[buf][0];
        const unsigned short* Bs = &sm[buf][GB_OFF];
        v16bf a0 = load_a(As + (size_t)(wave * 32) * GLDA,      GLDA, 0, lane);
        v16bf a1 = load_a(As + (size_t)(wave * 32 + 16) * GLDA, GLDA, 0, lane);
#pragma unroll
        for (int t = 0; t < 4; ++t) {
            v16bf b = load_bt(Bs + (size_t)(t * 16) * GLDA, GLDA, 0, lane);
            acc[t]     = wmma_bf16(a0, b, acc[t]);
            acc[4 + t] = wmma_bf16(a1, b, acc[4 + t]);
        }
        __syncthreads();
    }

    int half = lane >> 4, n = lane & 15;
    int rowBase = row0 + wave * 32;
#pragma unroll
    for (int t = 0; t < 4; ++t) {
        int col = col0 + t * 16 + n;
        float bv = bias ? bias[col] : 0.0f;
#pragma unroll
        for (int r = 0; r < 8; ++r) {
            C[(size_t)(rowBase + half * 8 + r) * Nc + col]      = acc[t][r] + bv;
            C[(size_t)(rowBase + 16 + half * 8 + r) * Nc + col] = acc[4 + t][r] + bv;
        }
    }
}

// ---------------------------------------------------------------------------
// Kernel 3: RMSNorm + RoPE (+ q-scale D^-0.5), qkv fp32 [B,N,3,H,D] ->
// q/k bf16 [B,H,N,D]; V additionally transposed -> bf16 [B,H,D,N] so the PV
// matmul gets K-contiguous B fragments. One wave per head per (b,n).
// ---------------------------------------------------------------------------
__global__ __launch_bounds__(384) void normrope(
        const float* __restrict__ qkv,
        const float* __restrict__ cosb, const float* __restrict__ sinb,
        const float* __restrict__ qw, const float* __restrict__ kw,
        unsigned short* __restrict__ qo, unsigned short* __restrict__ ko,
        unsigned short* __restrict__ vt) {
    int lane = threadIdx.x & 31;
    int h    = threadIdx.x >> 5;      // 0..11
    int bn   = blockIdx.x;            // b*N + n
    int n    = bn & (NN - 1);
    int b    = bn >> 11;

    const float* base = qkv + (size_t)bn * (3 * CC) + h * DD;
    int d0 = lane, d1 = lane + 32;
    float cs = cosb[n * (DD / 2) + lane];
    float sn = sinb[n * (DD / 2) + lane];
    size_t ob = (((size_t)b * HH + h) * NN + n) * DD;

    // ---- Q: rmsnorm + rope + scale ----
    {
        float x0 = base[d0], x1 = base[d1];
        float ss = x0 * x0 + x1 * x1;
#pragma unroll
        for (int m = 16; m >= 1; m >>= 1) ss += __shfl_xor(ss, m, 32);
        float rn = rsqrtf(ss * (1.0f / DD) + 1e-6f);
        x0 = x0 * rn * qw[d0];
        x1 = x1 * rn * qw[d1];
        qo[ob + d0] = f2bf((x0 * cs - x1 * sn) * 0.125f);   // fold D^-0.5
        qo[ob + d1] = f2bf((x1 * cs + x0 * sn) * 0.125f);
    }
    // ---- K: rmsnorm + rope ----
    {
        const float* kb = base + CC;
        float x0 = kb[d0], x1 = kb[d1];
        float ss = x0 * x0 + x1 * x1;
#pragma unroll
        for (int m = 16; m >= 1; m >>= 1) ss += __shfl_xor(ss, m, 32);
        float rn = rsqrtf(ss * (1.0f / DD) + 1e-6f);
        x0 = x0 * rn * kw[d0];
        x1 = x1 * rn * kw[d1];
        ko[ob + d0] = f2bf(x0 * cs - x1 * sn);
        ko[ob + d1] = f2bf(x1 * cs + x0 * sn);
    }
    // ---- V: convert + transpose to [B,H,D,N] ----
    {
        const float* vb = base + 2 * CC;
        size_t tb = (((size_t)b * HH + h) * DD) * NN + n;
        vt[tb + (size_t)d0 * NN] = f2bf(vb[d0]);
        vt[tb + (size_t)d1 * NN] = f2bf(vb[d1]);
    }
}

// ---------------------------------------------------------------------------
// Kernel 4: flash attention. One block = 4 waves = 64 queries of one (b,h);
// K tile (64x64) and V^T tile (64x64) double-buffered in LDS via async
// copies (ASYNCcnt) and shared by all waves. Online softmax in accumulator
// layout; P restaged through per-wave LDS into A-fragment layout for the
// PV WMMA. LDS tile rows padded to 72 elements (144 B).
// ---------------------------------------------------------------------------
#define FLD 72

__global__ __launch_bounds__(128) void flash_attn(
        const unsigned short* __restrict__ q,
        const unsigned short* __restrict__ k,
        const unsigned short* __restrict__ vt,
        unsigned short* __restrict__ o) {
    __shared__ unsigned short Kt[2][64 * FLD];
    __shared__ unsigned short Vt[2][64 * FLD];
    __shared__ unsigned short Pl[4][16 * 64];

    int tid  = threadIdx.x;
    int lane = tid & 31;
    int wave = tid >> 5;
    int bh   = blockIdx.x >> 5;       // 0..47
    int qb   = blockIdx.x & 31;       // 64-query block, 0..31

    const unsigned short* qbp = q  + ((size_t)bh * NN + qb * 64 + wave * 16) * DD;
    const unsigned short* kbp = k  + (size_t)bh * NN * DD;
    const unsigned short* vtb = vt + (size_t)bh * DD * NN;

    v16bf qa[2];
    qa[0] = load_a(qbp, DD, 0, lane);
    qa[1] = load_a(qbp, DD, 32, lane);

    v8f oacc[4];
    float rm[8], rs[8];
#pragma unroll
    for (int t = 0; t < 4; ++t) oacc[t] = vzero();
#pragma unroll
    for (int r = 0; r < 8; ++r) { rm[r] = -1e30f; rs[r] = 0.0f; }

    auto issue = [&](int kbase, int buf) {
#pragma unroll
        for (int i = 0; i < 8; ++i) {                  // 1024 16B chunks/stage
            int c = tid + i * 128;
            int isV = c >> 9;
            int idx = c & 511;
            int r = idx >> 3, c8 = idx & 7;
            if (isV)
                async_cp16((unsigned)(size_t)&Vt[buf][r * FLD + c8 * 8],
                           vtb + (size_t)r * NN + kbase + c8 * 8);
            else
                async_cp16((unsigned)(size_t)&Kt[buf][r * FLD + c8 * 8],
                           kbp + (size_t)(kbase + r) * DD + c8 * 8);
        }
    };

    unsigned short* Pw = &Pl[wave][0];
    int half = lane >> 4, nn = lane & 15;

    issue(0, 0);
    for (int st = 0; st < NN / 64; ++st) {
        int buf = st & 1;
        if (st + 1 < NN / 64) { issue((st + 1) * 64, buf ^ 1); S_WAIT_ASYNC8; }
        else                  { S_WAIT_ASYNC0; }
        __syncthreads();

        // ---- S = Q @ K^T  (16 x 64) ----
        v8f sc[4];
#pragma unroll
        for (int t = 0; t < 4; ++t) {
            sc[t] = vzero();
            const unsigned short* ktp = &Kt[buf][(t * 16) * FLD];
            sc[t] = wmma_bf16(qa[0], load_bt(ktp, FLD, 0, lane), sc[t]);
            sc[t] = wmma_bf16(qa[1], load_bt(ktp, FLD, 32, lane), sc[t]);
        }

        // ---- online softmax: row m = r + half*8 lives in VGPR r of 16 lanes
#pragma unroll
        for (int r = 0; r < 8; ++r) {
            float mx = sc[0][r];
#pragma unroll
            for (int t = 1; t < 4; ++t) mx = fmaxf(mx, sc[t][r]);
#pragma unroll
            for (int m = 8; m >= 1; m >>= 1) mx = fmaxf(mx, __shfl_xor(mx, m, 32));
            float nm   = fmaxf(rm[r], mx);
            float corr = __expf(rm[r] - nm);
            rm[r] = nm;
            float ps = 0.0f;
#pragma unroll
            for (int t = 0; t < 4; ++t) {
                float p = __expf(sc[t][r] - nm);
                sc[t][r] = p;
                ps += p;
            }
#pragma unroll
            for (int m = 8; m >= 1; m >>= 1) ps += __shfl_xor(ps, m, 32);
            rs[r] = rs[r] * corr + ps;
#pragma unroll
            for (int dt = 0; dt < 4; ++dt) oacc[dt][r] *= corr;
        }

        // ---- stage P (bf16) in LDS: C/D layout -> row-major 16x64 ----
#pragma unroll
        for (int t = 0; t < 4; ++t)
#pragma unroll
            for (int r = 0; r < 8; ++r)
                Pw[(half * 8 + r) * 64 + t * 16 + nn] = f2bf(sc[t][r]);
        __syncthreads();

        // ---- O += P @ V  (V^T tile: rows are d, K-contiguous) ----
#pragma unroll
        for (int c = 0; c < 2; ++c) {
            v16bf pa = load_a(Pw, 64, c * 32, lane);
#pragma unroll
            for (int dt = 0; dt < 4; ++dt) {
                v16bf bv = load_bt(&Vt[buf][(dt * 16) * FLD], FLD, c * 32, lane);
                oacc[dt] = wmma_bf16(pa, bv, oacc[dt]);
            }
        }
        __syncthreads();
    }

    // ---- epilogue: normalize, write bf16 to [B, N, h*D + d] ----
    int b = bh / HH, h = bh % HH;
#pragma unroll
    for (int r = 0; r < 8; ++r) {
        float inv  = 1.0f / rs[r];
        int   qrow = qb * 64 + wave * 16 + half * 8 + r;
        size_t rowb = ((size_t)b * NN + qrow) * CC + h * DD;
#pragma unroll
        for (int dt = 0; dt < 4; ++dt)
            o[rowb + dt * 16 + nn] = f2bf(oacc[dt][r] * inv);
    }
}

// ---------------------------------------------------------------------------
// Host launch
// ---------------------------------------------------------------------------
extern "C" void kernel_launch(void* const* d_in, const int* in_sizes, int n_in,
                              void* d_out, int out_size, void* d_ws, size_t ws_size,
                              hipStream_t stream) {
    const float* x      = (const float*)d_in[0];
    const float* cosb   = (const float*)d_in[1];
    const float* sinb   = (const float*)d_in[2];
    const float* qkv_w  = (const float*)d_in[3];
    const float* qnw    = (const float*)d_in[4];
    const float* knw    = (const float*)d_in[5];
    const float* proj_w = (const float*)d_in[6];
    const float* proj_b = (const float*)d_in[7];
    float* out = (float*)d_out;

    const size_t nX   = (size_t)BB * NN * CC;          // 6291456
    const size_t nWq  = (size_t)3 * CC * CC;           // 1769472
    const size_t nWp  = (size_t)CC * CC;               // 589824
    const size_t nQKV = (size_t)BB * NN * 3 * CC;      // 18874368
    const size_t nHD  = (size_t)BB * HH * NN * DD;     // 6291456

    unsigned short* x_bf     = (unsigned short*)d_ws;
    unsigned short* wqkv_bf  = x_bf + nX;
    unsigned short* wproj_bf = wqkv_bf + nWq;
    float*          qkv_f    = (float*)(wproj_bf + nWp);
    unsigned short* q_bf     = (unsigned short*)(qkv_f + nQKV);
    unsigned short* k_bf     = q_bf + nHD;
    unsigned short* vt_bf    = k_bf + nHD;
    unsigned short* attn_bf  = vt_bf + nHD;

    // 1) demote operands to bf16
    cvt_bf16<<<(int)((nX  + 255) / 256), 256, 0, stream>>>(x,      x_bf,     (int)nX);
    cvt_bf16<<<(int)((nWq + 255) / 256), 256, 0, stream>>>(qkv_w,  wqkv_bf,  (int)nWq);
    cvt_bf16<<<(int)((nWp + 255) / 256), 256, 0, stream>>>(proj_w, wproj_bf, (int)nWp);

    // 2) QKV projection: [8192,768] x [2304,768]^T -> fp32 [8192,2304]
    gemm_bf16<<<dim3(3 * CC / 64, BB * NN / 128), 128, 0, stream>>>(
        x_bf, wqkv_bf, qkv_f, nullptr, BB * NN, 3 * CC, CC);

    // 3) rmsnorm + rope + scale -> q/k bf16 [B,H,N,D], v^T bf16 [B,H,D,N]
    normrope<<<BB * NN, 384, 0, stream>>>(qkv_f, cosb, sinb, qnw, knw,
                                          q_bf, k_bf, vt_bf);

    // 4) flash attention -> bf16 [B,N,C]
    flash_attn<<<48 * (NN / 64), 128, 0, stream>>>(q_bf, k_bf, vt_bf, attn_bf);

    // 5) output projection + bias: [8192,768] x [768,768]^T -> fp32 out
    gemm_bf16<<<dim3(CC / 64, BB * NN / 128), 128, 0, stream>>>(
        attn_bf, wproj_bf, out, proj_b, BB * NN, CC, CC);
}
